// RopeAttention_16020228014145
// MI455X (gfx1250) — matrix operand order repaired and checked by reading, not compiled, not run on hardware
//
#include <hip/hip_runtime.h>

// ---------------------------------------------------------------------------
// RoPE causal attention for MI455X (gfx1250), bf16 WMMA pipeline.
//   B=4, T=2048, H=16, Dh=128, C=H*Dh=2048.
// All matmuls go through v_wmma_f32_16x16x32_bf16 (f32 accumulate).
// Direct global->LDS stagings use GLOBAL_LOAD_ASYNC_TO_LDS_B128 (ASYNCcnt)
// when the toolchain exposes the builtin; otherwise fall back to
// global_load_b128 + ds_store_b128.
// ---------------------------------------------------------------------------

typedef __bf16 bf16_t;
typedef __attribute__((ext_vector_type(16))) __bf16 bf16x16;
typedef __attribute__((ext_vector_type(8)))  __bf16 bf16x8;
typedef __attribute__((ext_vector_type(8)))  float  f32x8;

#define BATCH_  4
#define SEQ_    2048
#define NHEAD_  16
#define DHEAD_  128
#define DMODEL_ 2048      // NHEAD_*DHEAD_ == DMODEL_ here
#define MROWS_  (BATCH_*SEQ_)   // 8192

// ---- CDNA5 async global->LDS copy (16B per lane), guarded ------------------
// Builtin signature (from this toolchain's diagnostic + upstream clang):
//   void __builtin_amdgcn_global_load_async_to_lds_b128(
//       __global v4i* src, __local v4i* dst, imm int offset, imm int cpol);
#if defined(__HIP_DEVICE_COMPILE__) && __has_builtin(__builtin_amdgcn_global_load_async_to_lds_b128)
#define ASYNC_LDS 1
typedef int v4i_async __attribute__((vector_size(16)));
typedef __attribute__((address_space(1))) v4i_async* gbl_v4i;
typedef __attribute__((address_space(3))) v4i_async* lds_v4i;
#define ASYNC_COPY16(dstLds, srcGbl)                                          \
  __builtin_amdgcn_global_load_async_to_lds_b128((gbl_v4i)(srcGbl),           \
                                                 (lds_v4i)(dstLds), 0, 0)
#else
#define ASYNC_LDS 0
#endif

__device__ __forceinline__ void wait_async_zero() {
#if ASYNC_LDS
#if __has_builtin(__builtin_amdgcn_s_wait_asynccnt)
  __builtin_amdgcn_s_wait_asynccnt(0);
#else
  asm volatile("s_wait_asynccnt 0x0" ::: "memory");
#endif
#endif
}

__device__ __forceinline__ f32x8 wmma_bf16(bf16x16 a, bf16x16 b, f32x8 c) {
  return __builtin_amdgcn_wmma_f32_16x16x32_bf16(false, a, false, b,
                                                 (short)0, c, false, false);
}

__device__ __forceinline__ bf16x16 cat8(bf16x8 lo, bf16x8 hi) {
  bf16x16 r;
#pragma unroll
  for (int i = 0; i < 8; ++i) { r[i] = lo[i]; r[i + 8] = hi[i]; }
  return r;
}

// ------------------------------ pack kernels -------------------------------

__global__ void cvt_f32_bf16(const float* __restrict__ in,
                             bf16_t* __restrict__ out, int n) {
  int i = blockIdx.x * blockDim.x + threadIdx.x;
  if (i < n) out[i] = (bf16_t)in[i];
}

// W[n][c][d] (f32) -> Wb[c][n*128+d] (bf16):  c-major so the projection is a
// plain [8192,2048]x[2048,2048] GEMM.
__global__ void pack_wqkv(const float* __restrict__ in,
                          bf16_t* __restrict__ out) {
  int i = blockIdx.x * blockDim.x + threadIdx.x;   // over 16*2048*128 = 2^22
  int d = i & 127;
  int c = (i >> 7) & 2047;
  int n = i >> 18;
  out[(size_t)c * DMODEL_ + n * DHEAD_ + d] = (bf16_t)in[i];
}

// ------------------------------ WMMA GEMM ----------------------------------
// C[M,N] = A[M,K] * B[K,N] (+bias), bf16 in, f32 accumulate.
// Block: 256 thr = 8 waves, 128x128 tile. Wave (wrow 0..3, wcol 0..1) owns
// 32x64 = 2x4 WMMA tiles. K-step 32.  MODE: 0 = bf16 out (+bias),
// 1 = bf16 out (+bias, RoPE epilogue), 2 = f32 out (+bias).
template <int MODE>
__global__ void __launch_bounds__(256)
gemm_bf16(const bf16_t* __restrict__ A, const bf16_t* __restrict__ Bm,
          const float* __restrict__ bias, void* __restrict__ Cout,
          int Mdim, int Ndim, int Kdim) {
  __shared__ bf16_t As[128][40];   // row-major A tile  (+8 pad)
  __shared__ bf16_t Bt[128][40];   // transposed B tile (+8 pad): Bt[n][k]

  const int tid  = threadIdx.x;
  const int wave = tid >> 5, lane = tid & 31;
  const int half = lane >> 4, lr = lane & 15;
  const int wrow = wave >> 1, wcol = wave & 1;
  const int rowBase = blockIdx.y * 128;
  const int colBase = blockIdx.x * 128;

  f32x8 acc[2][4];
#pragma unroll
  for (int rt = 0; rt < 2; ++rt)
#pragma unroll
    for (int ct = 0; ct < 4; ++ct) acc[rt][ct] = (f32x8){};

  for (int k0 = 0; k0 < Kdim; k0 += 32) {
    // A tile: 128x32, each thread moves 16 contiguous elements (32B).
    {
      const int arow = tid >> 1, acol = (tid & 1) * 16;
      const bf16_t* gp = A + (size_t)(rowBase + arow) * Kdim + k0 + acol;
#if ASYNC_LDS
      ASYNC_COPY16(&As[arow][acol],     gp);
      ASYNC_COPY16(&As[arow][acol + 8], gp + 8);
#else
      bf16x8 v0 = *(const bf16x8*)gp;
      bf16x8 v1 = *(const bf16x8*)(gp + 8);
      *(bf16x8*)&As[arow][acol]     = v0;
      *(bf16x8*)&As[arow][acol + 8] = v1;
#endif
      if (k0 + 32 < Kdim) __builtin_prefetch(gp + 32, 0, 1);
    }
    // B tile transposed into Bt[n][k]: thread loads 16 n's of one k row.
    {
      const int kr = tid >> 3, nc = (tid & 7) * 16;
      const bf16_t* gp = Bm + (size_t)(k0 + kr) * Ndim + colBase + nc;
      bf16x8 v0 = *(const bf16x8*)gp;
      bf16x8 v1 = *(const bf16x8*)(gp + 8);
#pragma unroll
      for (int i = 0; i < 8; ++i) {
        Bt[nc + i][kr]     = v0[i];
        Bt[nc + 8 + i][kr] = v1[i];
      }
      if (k0 + 32 < Kdim) __builtin_prefetch(gp + (size_t)32 * Ndim, 0, 1);
    }
    wait_async_zero();
    __syncthreads();

    // A fragments: lane<16 holds K 0-7 & 16-23; lane>=16 holds K 8-15 & 24-31.
    bf16x16 af[2];
#pragma unroll
    for (int rt = 0; rt < 2; ++rt) {
      const int r = wrow * 32 + rt * 16 + lr;
      bf16x8 lo = *(const bf16x8*)&As[r][half * 8];
      bf16x8 hi = *(const bf16x8*)&As[r][16 + half * 8];
      af[rt] = cat8(lo, hi);
    }
    // B fragments: lane<16 holds K 0-7 & 16-23; lane>=16 holds K 8-15 & 24-31 for column n = base+lr.
    bf16x16 bfr[4];
#pragma unroll
    for (int ct = 0; ct < 4; ++ct) {
      const int nn = wcol * 64 + ct * 16 + lr;
      const bf16_t* p = &Bt[nn][0];
      bfr[ct] = cat8(*(const bf16x8*)(p + half * 8), *(const bf16x8*)(p + 16 + half * 8));
    }
#pragma unroll
    for (int rt = 0; rt < 2; ++rt)
#pragma unroll
      for (int ct = 0; ct < 4; ++ct)
        acc[rt][ct] = wmma_bf16(af[rt], bfr[ct], acc[rt][ct]);
    __syncthreads();
  }

  // Epilogue. C/D layout: VGPR vi <-> row vi (+8 for upper half), col = lr.
#pragma unroll
  for (int rt = 0; rt < 2; ++rt) {
#pragma unroll
    for (int ct = 0; ct < 4; ++ct) {
      const int col = colBase + wcol * 64 + ct * 16 + lr;
      const float bv = bias[col];
#pragma unroll
      for (int vi = 0; vi < 8; ++vi) {
        const int row = rowBase + wrow * 32 + rt * 16 + vi + half * 8;
        float v = acc[rt][ct][vi] + bv;
        if (MODE == 1) {
          // RoPE: pair (2j, 2j+1) are adjacent columns -> adjacent lanes.
          float p = __shfl_xor(v, 1, 32);
          const int d = col & (DHEAD_ - 1);
          const int t = row & (SEQ_ - 1);
          const int j = d >> 1;
          float inv = __powf(10000.0f, -(float)(2 * j) * (1.0f / DHEAD_));
          float ang = (float)t * inv;
          float cs = __cosf(ang), sn = __sinf(ang);
          v = (d & 1) ? (p * sn + v * cs) : (v * cs - p * sn);
        }
        if (MODE == 2)
          ((float*)Cout)[(size_t)row * Ndim + col] = v;
        else
          ((bf16_t*)Cout)[(size_t)row * Ndim + col] = (bf16_t)v;
      }
    }
  }
}

// --------------------------- flash attention -------------------------------
// One block per (q-tile of 128, head, batch). 8 waves; wave w owns 16 query
// rows. K block (64x128) and V^T (128x64) staged in LDS; P re-shaped via a
// per-wave LDS tile from C-layout to A-fragment layout.
__global__ void __launch_bounds__(256)
flash_rope_attn(const bf16_t* __restrict__ Q, const bf16_t* __restrict__ K,
                const bf16_t* __restrict__ V, bf16_t* __restrict__ Z) {
  __shared__ bf16_t Kl[64][DHEAD_ + 8];   // K rows, d contiguous
  __shared__ bf16_t Vt[DHEAD_][64 + 8];   // V transposed: Vt[d][k]
  __shared__ bf16_t Pl[8][16][64];        // per-wave P tile

  const int tid  = threadIdx.x;
  const int wave = tid >> 5, lane = tid & 31;
  const int half = lane >> 4, lr = lane & 15;
  const int qbase = blockIdx.x * 128;
  const int head  = blockIdx.y;
  const int b     = blockIdx.z;
  const int q0    = qbase + wave * 16;

  const size_t batchOff = (size_t)b * SEQ_ * DMODEL_;
  const size_t headOff  = (size_t)head * DHEAD_;

  // Q fragments for this wave's 16 rows (4 K-chunks of 32 along d).
  bf16x16 qf[4];
#pragma unroll
  for (int dk = 0; dk < 4; ++dk) {
    const bf16_t* p = Q + batchOff + (size_t)(q0 + lr) * DMODEL_ + headOff + dk * 32;
    qf[dk] = cat8(*(const bf16x8*)(p + half * 8),
                  *(const bf16x8*)(p + 16 + half * 8));
  }

  f32x8 z[8];
#pragma unroll
  for (int dt = 0; dt < 8; ++dt) z[dt] = (f32x8){};
  float mstate[8], lstate[8];
#pragma unroll
  for (int i = 0; i < 8; ++i) { mstate[i] = -3.0e38f; lstate[i] = 0.0f; }

  const int nkb = (qbase + 128) / 64;   // causal bound on key blocks
  for (int kb = 0; kb < nkb; ++kb) {
    const int kbase = kb * 64;
    // Cooperative K / V^T staging: 2 chunks of 16 elems per thread.
#pragma unroll
    for (int cc = 0; cc < 2; ++cc) {
      const int c  = tid * 2 + cc;        // 0..511
      const int rr = c >> 3;              // key row 0..63
      const int c0 = (c & 7) * 16;        // d offset
      const bf16_t* kp = K + batchOff + (size_t)(kbase + rr) * DMODEL_ + headOff + c0;
#if ASYNC_LDS
      ASYNC_COPY16(&Kl[rr][c0],     kp);
      ASYNC_COPY16(&Kl[rr][c0 + 8], kp + 8);
#else
      bf16x8 k0v = *(const bf16x8*)kp;
      bf16x8 k1v = *(const bf16x8*)(kp + 8);
      *(bf16x8*)&Kl[rr][c0]     = k0v;
      *(bf16x8*)&Kl[rr][c0 + 8] = k1v;
#endif
      const bf16_t* vp = V + batchOff + (size_t)(kbase + rr) * DMODEL_ + headOff + c0;
      bf16x8 v0v = *(const bf16x8*)vp;
      bf16x8 v1v = *(const bf16x8*)(vp + 8);
#pragma unroll
      for (int i = 0; i < 8; ++i) {
        Vt[c0 + i][rr]     = v0v[i];
        Vt[c0 + 8 + i][rr] = v1v[i];
      }
    }
    wait_async_zero();
    __syncthreads();

    // S = Q K^T over 4 key tiles x 4 d-chunks.
    f32x8 s[4];
#pragma unroll
    for (int ct = 0; ct < 4; ++ct) {
      f32x8 a = (f32x8){};
#pragma unroll
      for (int dk = 0; dk < 4; ++dk) {
        const bf16_t* p = &Kl[ct * 16 + lr][dk * 32];
        bf16x16 kf = cat8(*(const bf16x8*)(p + half * 8), *(const bf16x8*)(p + 16 + half * 8));
        a = wmma_bf16(qf[dk], kf, a);
      }
      s[ct] = a;
    }

    // scale + causal mask
    const float scale = 0.08838834764831845f;   // 1/sqrt(128)
#pragma unroll
    for (int ct = 0; ct < 4; ++ct) {
      const int kg = kbase + ct * 16 + lr;
#pragma unroll
      for (int vi = 0; vi < 8; ++vi) {
        const int qg = q0 + vi + half * 8;
        const float sv = s[ct][vi] * scale;
        s[ct][vi] = (kg > qg) ? -3.0e38f : sv;
      }
    }

    // Online softmax; row lives in a 16-lane half -> butterfly stays in-half.
#pragma unroll
    for (int vi = 0; vi < 8; ++vi) {
      float mx = fmaxf(fmaxf(s[0][vi], s[1][vi]), fmaxf(s[2][vi], s[3][vi]));
#pragma unroll
      for (int off = 1; off < 16; off <<= 1) mx = fmaxf(mx, __shfl_xor(mx, off, 32));
      const float mn = fmaxf(mstate[vi], mx);
      const float alpha = __expf(mstate[vi] - mn);
      mstate[vi] = mn;
#pragma unroll
      for (int dt = 0; dt < 8; ++dt) z[dt][vi] *= alpha;
      float rs = 0.0f;
#pragma unroll
      for (int ct = 0; ct < 4; ++ct) {
        const float pv = __expf(s[ct][vi] - mn);
        s[ct][vi] = pv;
        rs += pv;
      }
#pragma unroll
      for (int off = 1; off < 16; off <<= 1) rs += __shfl_xor(rs, off, 32);
      lstate[vi] = lstate[vi] * alpha + rs;
    }

    // C-layout -> A-layout via per-wave LDS tile (same-wave LDS is in-order).
#pragma unroll
    for (int ct = 0; ct < 4; ++ct)
#pragma unroll
      for (int vi = 0; vi < 8; ++vi)
        Pl[wave][vi + half * 8][ct * 16 + lr] = (bf16_t)s[ct][vi];

    // z += P * V
#pragma unroll
    for (int kc = 0; kc < 2; ++kc) {
      const bf16_t* pp = &Pl[wave][lr][kc * 32];
      bf16x16 pf = cat8(*(const bf16x8*)(pp + half * 8),
                        *(const bf16x8*)(pp + 16 + half * 8));
#pragma unroll
      for (int dt = 0; dt < 8; ++dt) {
        const bf16_t* vp = &Vt[dt * 16 + lr][kc * 32];
        bf16x16 vf = cat8(*(const bf16x8*)(vp + half * 8), *(const bf16x8*)(vp + 16 + half * 8));
        z[dt] = wmma_bf16(pf, vf, z[dt]);
      }
    }
    __syncthreads();
  }

  // Normalize and store Z (bf16) for the output projection.
#pragma unroll
  for (int vi = 0; vi < 8; ++vi) {
    const float inv = 1.0f / lstate[vi];
    const int q = q0 + vi + half * 8;
#pragma unroll
    for (int dt = 0; dt < 8; ++dt) {
      const int d = dt * 16 + lr;
      Z[batchOff + (size_t)q * DMODEL_ + headOff + d] = (bf16_t)(z[dt][vi] * inv);
    }
  }
}

// ------------------------------- launch ------------------------------------

extern "C" void kernel_launch(void* const* d_in, const int* in_sizes, int n_in,
                              void* d_out, int out_size, void* d_ws, size_t ws_size,
                              hipStream_t stream) {
  (void)in_sizes; (void)n_in; (void)out_size; (void)ws_size;
  const float* x   = (const float*)d_in[0];
  const float* W_Q = (const float*)d_in[1];
  const float* b_Q = (const float*)d_in[2];
  const float* W_K = (const float*)d_in[3];
  const float* b_K = (const float*)d_in[4];
  const float* W_V = (const float*)d_in[5];
  const float* b_V = (const float*)d_in[6];
  const float* W_O = (const float*)d_in[7];
  const float* b_O = (const float*)d_in[8];
  float* out = (float*)d_out;

  const size_t SZ = (size_t)MROWS_ * DMODEL_;   // 16,777,216 elems
  const size_t WZ = (size_t)DMODEL_ * DMODEL_;  //  4,194,304 elems
  bf16_t* Xb  = (bf16_t*)d_ws;
  bf16_t* Qb  = Xb + SZ;
  bf16_t* Kb  = Qb + SZ;
  bf16_t* Vb  = Kb + SZ;
  bf16_t* Zb  = Vb + SZ;
  bf16_t* Wqb = Zb + SZ;
  bf16_t* Wkb = Wqb + WZ;
  bf16_t* Wvb = Wkb + WZ;
  bf16_t* Wob = Wvb + WZ;

  // 1) pack / convert
  cvt_f32_bf16<<<(int)(SZ / 256), 256, 0, stream>>>(x, Xb, (int)SZ);
  pack_wqkv<<<(int)(WZ / 256), 256, 0, stream>>>(W_Q, Wqb);
  pack_wqkv<<<(int)(WZ / 256), 256, 0, stream>>>(W_K, Wkb);
  pack_wqkv<<<(int)(WZ / 256), 256, 0, stream>>>(W_V, Wvb);
  cvt_f32_bf16<<<(int)(WZ / 256), 256, 0, stream>>>(W_O, Wob, (int)WZ);

  // 2) QKV projections (RoPE fused into Q/K epilogues)
  dim3 gg(DMODEL_ / 128, MROWS_ / 128);   // (16, 64)
  gemm_bf16<1><<<gg, 256, 0, stream>>>(Xb, Wqb, b_Q, Qb, MROWS_, DMODEL_, DMODEL_);
  gemm_bf16<1><<<gg, 256, 0, stream>>>(Xb, Wkb, b_K, Kb, MROWS_, DMODEL_, DMODEL_);
  gemm_bf16<0><<<gg, 256, 0, stream>>>(Xb, Wvb, b_V, Vb, MROWS_, DMODEL_, DMODEL_);

  // 3) causal flash attention
  flash_rope_attn<<<dim3(SEQ_ / 128, NHEAD_, BATCH_), 256, 0, stream>>>(Qb, Kb, Vb, Zb);

  // 4) output projection -> fp32
  gemm_bf16<2><<<gg, 256, 0, stream>>>(Zb, Wob, b_O, out, MROWS_, DMODEL_, DMODEL_);
}
